// OutProductMean_36618891165708
// MI455X (gfx1250) — matrix operand order repaired and checked by reading, not compiled, not run on hardware
//
#include <hip/hip_runtime.h>
#include <hip/hip_bf16.h>
#include <stddef.h>
#include <stdint.h>

// ---------------------------------------------------------------------------
// OuterProductMean for MI455X (gfx1250), wave32 + v_wmma_f32_16x16x32_bf16
// + Tensor Data Mover (tensor_load_to_lds) double-buffered B-tile staging.
//
// Shapes: S=512 seq, R=256 res, C_M=256, C_P=32, C_Z=128.
// Dominant GEMM: O[(i,d),(j,e)] = sum_s L[s,i,d]*R[s,j,e]  (8192x8192x512)
// fused with Z = O_flat(1024) x Wo^T(1024x128) per (i,j) pair.
// ---------------------------------------------------------------------------

typedef __attribute__((ext_vector_type(16))) __bf16 bf16x16;
typedef __attribute__((ext_vector_type(8)))  float  f32x8;
typedef __attribute__((ext_vector_type(4)))  unsigned int u32x4;
typedef __attribute__((ext_vector_type(8)))  int  i32x8;
typedef __attribute__((ext_vector_type(4)))  int  i32x4;

union Frag32 {                 // one 16x32(A) / 32x16(B) bf16 WMMA operand per lane
    bf16x16 bf;
    uint4   q[2];              // 2 x 16B
    unsigned short us[16];
};

__device__ __forceinline__ unsigned short f2bf(float f) {
    unsigned u = __float_as_uint(f);
    unsigned r = u + 0x7FFFu + ((u >> 16) & 1u);   // round-to-nearest-even
    return (unsigned short)(r >> 16);
}

#if __has_builtin(__builtin_amdgcn_tensor_load_to_lds) && \
    __has_builtin(__builtin_amdgcn_s_wait_tensorcnt)
#define HAVE_TDM 1
#endif

#ifdef HAVE_TDM
// 1D TDM copy: ndw dwords, contiguous global -> contiguous LDS.
// D# per CDNA5 ISA ch.8: group0 {count=1, lds_addr, global_addr[56:0], type=2},
// group1 {data_size=4B, tensor_dim0 = tile_dim0 = ndw, stride0 = ndw}.
__device__ __forceinline__ void tdm_load_1d(unsigned lds_addr, const void* gaddr,
                                            unsigned ndw) {
    unsigned long long ga = (unsigned long long)(uintptr_t)gaddr;
    u32x4 g0 = { 1u,                                    // count=1, is_restore=0
                 lds_addr,                              // bits 63:32
                 (unsigned)(ga & 0xffffffffu),          // global_addr[31:0]
                 (unsigned)((ga >> 32) & 0x01ffffffu)   // global_addr[56:32]
                   | (2u << 30) };                      // type=2 ("image")
    i32x8 g1 = { (int)(2u << 16),                       // data_size=2 (4 bytes)
                 (int)((ndw & 0xffffu) << 16),          // tensor_dim0[15:0] @ bits 63:48
                 (int)(((ndw >> 16) & 0xffffu) | (1u << 16)), // td0[31:16], tensor_dim1=1
                 (int)((ndw & 0xffffu) << 16),          // tile_dim0 @ bits 127:112
                 0,                                     // tile_dim1/2 = 0 (1D)
                 (int)ndw,                              // tensor_dim0_stride[31:0]
                 0, 0 };
    i32x4 z4 = {0, 0, 0, 0};
#if __clang_major__ >= 23
    i32x8 z8 = {0, 0, 0, 0, 0, 0, 0, 0};
    __builtin_amdgcn_tensor_load_to_lds(g0, g1, z4, z4, z8, 0);
#else
    __builtin_amdgcn_tensor_load_to_lds(g0, g1, z4, z4, 0);
#endif
}
#endif

// Workspace layout (bytes)
#define LPACK_OFF   0u                      // 16*512*32*16 bf16 = 8 MB (A-pack of L)
#define RPACK_OFF   (8u << 20)              // 8 MB (B-pack of R)
#define WAPK_OFF    (16u << 20)             // 8*2*32*16 bf16 = 16 KB
#define WBPK_OFF    ((16u << 20) + (16u << 10))
#define WOPK_OFF    ((16u << 20) + (32u << 10))   // 32*8*32*16 bf16 = 256 KB
#define NORM_OFF    ((16u << 20) + (32u << 10) + (256u << 10))  // 256*256 f32

// ---------------------------------------------------------------------------
// Kernel 0: pre-pack weights into WMMA B-fragment lane layout (bf16).
// B-frag (32x16, K x N): lane l holds column n = nt*16 + (l&15),
//   halves j=0..15 are K = kt*32 + ((l&16)?8:0) + (j&7) + ((j&8)?16:0) (32B/lane).
// ---------------------------------------------------------------------------
__global__ __launch_bounds__(256) void k0_pack(const float* __restrict__ Wa,
                                               const float* __restrict__ Wb,
                                               const float* __restrict__ Wo,
                                               unsigned short* WaPk,
                                               unsigned short* WbPk,
                                               unsigned short* WoPk) {
    int idx = blockIdx.x * 256 + threadIdx.x;
    if (idx < 16384) {                         // Wa / Wb : 32x256, K=256 (8 kt), N=32 (2 nt)
        const float* W = (idx < 8192) ? Wa : Wb;
        unsigned short* P = (idx < 8192) ? WaPk : WbPk;
        int t = idx & 8191;
        int kt = t >> 10, rem = t & 1023;
        int nt = rem >> 9, l = (rem >> 4) & 31, j = rem & 15;
        int n = nt * 16 + (l & 15);
        int k = kt * 32 + ((l & 16) ? 8 : 0) + (j & 7) + ((j & 8) ? 16 : 0);
        P[t] = f2bf(W[n * 256 + k]);           // B[k][n] = W[n][k]
    } else {
        int t = idx - 16384;                   // Wo : 128x1024, K=1024 (32 kt), N=128 (8 nt)
        if (t < 131072) {
            int kt = t >> 12, rem = t & 4095;
            int nt = rem >> 9, l = (rem >> 4) & 31, j = rem & 15;
            int z = nt * 16 + (l & 15);
            int k = kt * 32 + ((l & 16) ? 8 : 0) + (j & 7) + ((j & 8) ? 16 : 0);
            WoPk[t] = f2bf(Wo[z * 1024 + k]);
        }
    }
}

// ---------------------------------------------------------------------------
// Kernel: norm[i][j] = sum_s mask[s,i]*mask[s,j]     (256x256, K=512)
// ---------------------------------------------------------------------------
__global__ __launch_bounds__(256) void k_norm(const float* __restrict__ mask,
                                              float* __restrict__ normArr) {
    int i = blockIdx.x, j = threadIdx.x;
    float acc = 0.f;
    for (int s = 0; s < 512; ++s)
        acc += mask[s * 256 + i] * mask[s * 256 + j];
    normArr[i * 256 + j] = acc;
}

// ---------------------------------------------------------------------------
// Kernel 1: LayerNorm (f32) + bf16 WMMA projections + bias + mask, then
// repack results straight into the big GEMM's fragment layouts:
//   Lpack[ktS][mt=i*2+h][lane][16]  (A-frag layout, M=(i,d), K=s)
//   Rpack[ktS][nt=i*2+h][lane][16]  (B-frag layout, N=(j,e), K=s)
// One WG = one residue i x 32 sequence rows.
// ---------------------------------------------------------------------------
__global__ __launch_bounds__(256) void k1_ln_proj(const float* __restrict__ M,
                                                  const float* __restrict__ M_mask,
                                                  const float* __restrict__ ln_g,
                                                  const float* __restrict__ ln_b,
                                                  const unsigned short* __restrict__ WaPk,
                                                  const unsigned short* __restrict__ WbPk,
                                                  const float* __restrict__ ba,
                                                  const float* __restrict__ bb,
                                                  unsigned short* __restrict__ Lpack,
                                                  unsigned short* __restrict__ Rpack) {
    const int i = blockIdx.x, sblk = blockIdx.y;
    const int tid = threadIdx.x, w = tid >> 5, lane = tid & 31;
    const int hi = (lane >> 4) & 1;
    const int s0 = sblk * 32;

    __shared__ float          Mlds[32][256];        // raw f32 rows
    __shared__ unsigned short MnL[32][256];         // normalized bf16
    __shared__ unsigned short Llds[32][32];
    __shared__ unsigned short Rlds[32][32];
    __shared__ float glds[256], blds[256], maskv[32];

    glds[tid] = ln_g[tid];
    blds[tid] = ln_b[tid];
    if (tid < 32) maskv[tid] = M_mask[(size_t)(s0 + tid) * 256 + i];
    for (int r = 0; r < 32; ++r)
        Mlds[r][tid] = M[(size_t)(s0 + r) * 65536 + (size_t)i * 256 + tid];
    __syncthreads();

    // LayerNorm: wave w handles rows 4w..4w+3; 32-lane reduction of 256 values
    for (int rr = 0; rr < 4; ++rr) {
        int row = w * 4 + rr;
        float s1 = 0.f, s2 = 0.f;
        for (int k = 0; k < 8; ++k) {
            float x = Mlds[row][lane + 32 * k];
            s1 += x; s2 += x * x;
        }
        for (int off = 16; off; off >>= 1) {
            s1 += __shfl_xor(s1, off, 32);
            s2 += __shfl_xor(s2, off, 32);
        }
        float mu   = s1 * (1.f / 256.f);
        float var  = s2 * (1.f / 256.f) - mu * mu;
        float rstd = rsqrtf(var + 1e-5f);
        for (int k = 0; k < 8; ++k) {
            int c = lane + 32 * k;
            MnL[row][c] = f2bf((Mlds[row][c] - mu) * rstd * glds[c] + blds[c]);
        }
    }
    __syncthreads();

    // Projections via WMMA: 8 output frags, one per wave.
    // w bits: [0]=s-tile, [1]=proj (a/b), [2]=n-half
    const int st = w & 1, pr = (w >> 1) & 1, nt = (w >> 2) & 1;
    const unsigned short* WPk = pr ? WbPk : WaPk;
    const int m = st * 16 + (lane & 15);
    f32x8 acc = {};
    for (int kt = 0; kt < 8; ++kt) {
        Frag32 fa, fb;
        int k0c = kt * 32 + hi * 8;                 // A layout: h[0..7]=K k0.., h[8..15]=K k0+16..
        fa.q[0] = *(const uint4*)&MnL[m][k0c];
        fa.q[1] = *(const uint4*)&MnL[m][k0c + 16];
        const unsigned short* bp = WPk + ((size_t)(kt * 2 + nt) * 32 + lane) * 16;
        fb.q[0] = *(const uint4*)bp;
        fb.q[1] = *(const uint4*)(bp + 8);
        acc = __builtin_amdgcn_wmma_f32_16x16x32_bf16(false, fa.bf, false, fb.bf,
                                                      (short)0, acc, false, false);
    }
    const int d = nt * 16 + (lane & 15);
    const float bval = (pr ? bb : ba)[d];
    unsigned short (*OutL)[32] = pr ? Rlds : Llds;
    for (int r = 0; r < 8; ++r) {
        int sl = st * 16 + r + hi * 8;              // C layout: vgpr r -> M=r (+8 in hi half)
        OutL[sl][d] = f2bf((acc[r] + bval) * maskv[sl]);
    }
    __syncthreads();

    // Repack to global fragment layouts (waves 0-3 only)
    if (w < 4) {
        const int h = w & 1;
        Frag32 t;
        unsigned short* dst;
        if (w < 2) {                                 // L -> A-pack: m-row = d, halves = s
            int dd = h * 16 + (lane & 15);
            int k0 = hi * 8;
            for (int j = 0; j < 8; ++j) t.us[j]     = Llds[k0 + j][dd];
            for (int j = 0; j < 8; ++j) t.us[8 + j] = Llds[k0 + 16 + j][dd];
            dst = Lpack + ((size_t)(sblk * 512 + (i * 2 + h)) * 32 + lane) * 16;
        } else {                                     // R -> B-pack: n-col = e, halves = s
            int ee = h * 16 + (lane & 15);
            int sb = hi * 8;
            for (int j = 0; j < 8; ++j) t.us[j]     = Rlds[sb + j][ee];
            for (int j = 0; j < 8; ++j) t.us[8 + j] = Rlds[sb + 16 + j][ee];
            dst = Rpack + ((size_t)(sblk * 512 + (i * 2 + h)) * 32 + lane) * 16;
        }
        *(uint4*)dst       = t.q[0];
        *(uint4*)(dst + 8) = t.q[1];
    }
}

// ---------------------------------------------------------------------------
// Kernel 3: big GEMM (128x128 O-tile per WG, K=512) fused with Wo projection
// (16 pairs x 1024 x 128) and mask-norm divide. B-tiles are contiguous 8 KB
// slabs of Rpack -> staged by the Tensor Data Mover, double-buffered.
// ---------------------------------------------------------------------------
__global__ __launch_bounds__(256) void k3_gemm(const unsigned short* __restrict__ Lpack,
                                               const unsigned short* __restrict__ Rpack,
                                               const unsigned short* __restrict__ WoPk,
                                               const float* __restrict__ normArr,
                                               const float* __restrict__ bo,
                                               float* __restrict__ out) {
    const int it = blockIdx.x, jt = blockIdx.y;        // 4 residues i / j each
    const int tid = threadIdx.x, w = tid >> 5, lane = tid & 31;
    const int hi = (lane >> 4) & 1;

    __shared__ unsigned short Btile[2][8 * 32 * 16];   // 2 x 8 KB K-slabs of R
    __shared__ unsigned short OL[16][1024];            // 32 KB (O-tile, bf16, per pair)
    __shared__ float normL[16];

#ifdef HAVE_TDM
    // TDM: one 8 KB descriptor per K-slab, issued by wave 0, TENSORcnt-tracked.
    if (w == 0) {
        tdm_load_1d((unsigned)(uintptr_t)&Btile[0][0],
                    Rpack + (size_t)(0 * 512 + jt * 8) * 512, 2048);
        tdm_load_1d((unsigned)(uintptr_t)&Btile[1][0],
                    Rpack + (size_t)(1 * 512 + jt * 8) * 512, 2048);
    }
#endif

    f32x8 acc[8] = {};
    for (int kt = 0; kt < 16; ++kt) {
        const int cur = kt & 1;
#ifdef HAVE_TDM
        if (w == 0) {                                  // retire current slab
            if (kt < 15) __builtin_amdgcn_s_wait_tensorcnt(1);
            else         __builtin_amdgcn_s_wait_tensorcnt(0);
        }
        __syncthreads();
#else
        __syncthreads();
        {   // fallback: cooperative stage of B-tile: 256 lanes x 32 B
            const unsigned short* src =
                Rpack + ((size_t)(kt * 512 + jt * 8 + w) * 32 + lane) * 16;
            unsigned short* dst = &Btile[cur][(w * 32 + lane) * 16];
            *(uint4*)dst       = *(const uint4*)src;
            *(uint4*)(dst + 8) = *(const uint4*)(src + 8);
        }
        __syncthreads();
#endif
        Frag32 fa;                                     // A-frag: wave w owns m-rows 16w..16w+15
        const unsigned short* ap =
            Lpack + ((size_t)(kt * 512 + it * 8 + w) * 32 + lane) * 16;
        fa.q[0] = *(const uint4*)ap;
        fa.q[1] = *(const uint4*)(ap + 8);
        __builtin_prefetch(ap + (size_t)512 * 32 * 16, 0, 1);   // next K-slab of A
        for (int nt = 0; nt < 8; ++nt) {
            Frag32 fb;
            const unsigned short* bp = &Btile[cur][(nt * 32 + lane) * 16];
            fb.q[0] = *(const uint4*)bp;
            fb.q[1] = *(const uint4*)(bp + 8);
            acc[nt] = __builtin_amdgcn_wmma_f32_16x16x32_bf16(false, fa.bf, false, fb.bf,
                                                              (short)0, acc[nt], false, false);
        }
#ifdef HAVE_TDM
        __syncthreads();                               // all readers done with Btile[cur]
        if (w == 0 && kt + 2 < 16)                     // refill it for slab kt+2
            tdm_load_1d((unsigned)(uintptr_t)&Btile[cur][0],
                        Rpack + (size_t)((kt + 2) * 512 + jt * 8) * 512, 2048);
#endif
    }
    __syncthreads();

    // Scatter O-tile into per-pair flat vectors OL[p][d*32+e] (bf16),
    // laid out so the second GEMM's A-frags are contiguous 16B LDS reads.
    const int il = w >> 1, db = (w & 1) * 16;
    for (int nt = 0; nt < 8; ++nt) {
        int jl = nt >> 1, eb = (nt & 1) * 16;
        int p = il * 4 + jl;
        int e = eb + (lane & 15);
        for (int r = 0; r < 8; ++r) {
            int dd = db + r + hi * 8;
            OL[p][dd * 32 + e] = f2bf(acc[nt][r]);
        }
    }
    if (tid < 16) {
        int p = tid, gi = it * 4 + (p >> 2), gj = jt * 4 + (p & 3);
        normL[p] = 1.0f / (1e-3f + normArr[gi * 256 + gj]);
    }
    __syncthreads();

    // Second GEMM: Z(16x128) = OL(16x1024) x WoPk(1024x128); wave w owns z-block w.
    f32x8 acc2 = {};
    const int p0 = lane & 15;
    for (int kt2 = 0; kt2 < 32; ++kt2) {
        Frag32 fa, fb;
        const unsigned short* ap = &OL[p0][kt2 * 32 + hi * 8];
        fa.q[0] = *(const uint4*)ap;
        fa.q[1] = *(const uint4*)(ap + 16);            // A layout: second half at K+16
        const unsigned short* bp = WoPk + ((size_t)(kt2 * 8 + w) * 32 + lane) * 16;
        fb.q[0] = *(const uint4*)bp;
        fb.q[1] = *(const uint4*)(bp + 8);
        acc2 = __builtin_amdgcn_wmma_f32_16x16x32_bf16(false, fa.bf, false, fb.bf,
                                                       (short)0, acc2, false, false);
    }
    const int z = w * 16 + (lane & 15);
    const float bz = bo[z];
    for (int r = 0; r < 8; ++r) {
        int p  = r + hi * 8;
        int gi = it * 4 + (p >> 2), gj = jt * 4 + (p & 3);
        out[((size_t)gi * 256 + gj) * 128 + z] = (acc2[r] + bz) * normL[p];
    }
}

// ---------------------------------------------------------------------------
extern "C" void kernel_launch(void* const* d_in, const int* in_sizes, int n_in,
                              void* d_out, int out_size, void* d_ws, size_t ws_size,
                              hipStream_t stream) {
    const float* M      = (const float*)d_in[0];   // [1,512,256,256]
    const float* M_mask = (const float*)d_in[1];   // [1,512,256]
    // d_in[2] = Z_raw (unused by the reference math)
    const float* ln_g   = (const float*)d_in[3];
    const float* ln_b   = (const float*)d_in[4];
    const float* Wa     = (const float*)d_in[5];   // [32,256]
    const float* ba     = (const float*)d_in[6];
    const float* Wb     = (const float*)d_in[7];   // [32,256]
    const float* bb     = (const float*)d_in[8];
    const float* Wo     = (const float*)d_in[9];   // [128,1024]
    const float* bo     = (const float*)d_in[10];
    float* out = (float*)d_out;

    char* ws = (char*)d_ws;
    unsigned short* Lpack = (unsigned short*)(ws + LPACK_OFF);
    unsigned short* Rpack = (unsigned short*)(ws + RPACK_OFF);
    unsigned short* WaPk  = (unsigned short*)(ws + WAPK_OFF);
    unsigned short* WbPk  = (unsigned short*)(ws + WBPK_OFF);
    unsigned short* WoPk  = (unsigned short*)(ws + WOPK_OFF);
    float*          normA = (float*)(ws + NORM_OFF);

    k0_pack<<<576, 256, 0, stream>>>(Wa, Wb, Wo, WaPk, WbPk, WoPk);
    k_norm<<<256, 256, 0, stream>>>(M_mask, normA);
    k1_ln_proj<<<dim3(256, 16), 256, 0, stream>>>(M, M_mask, ln_g, ln_b,
                                                  WaPk, WbPk, ba, bb, Lpack, Rpack);
    k3_gemm<<<dim3(64, 64), 256, 0, stream>>>(Lpack, Rpack, WoPk, normA, bo, out);
}
